// OrthogonalBivectorBlock_25348896981073
// MI455X (gfx1250) — compile-verified, matrix operand order checked by reading, not run on hardware
//
#include <hip/hip_runtime.h>
#include <hip/hip_bf16.h>
#include <math.h>

typedef __attribute__((ext_vector_type(16))) _Float16 v16h;
typedef __attribute__((ext_vector_type(8)))  float    v8f;
typedef int v4i_vs __attribute__((vector_size(16)));

#define TOKENS 4096
#define LSEQ   2048
#define DDIM   512
#define NCHUNK 32
#define CHUNK  64
#define PI_F   3.14159265358979323846f

#if defined(__AMDGCN__) && __has_builtin(__builtin_amdgcn_global_load_async_to_lds_b128)
#define HAVE_ASYNC_LDS 1
#else
#define HAVE_ASYNC_LDS 0
#endif

__device__ __forceinline__ float gelu_exact(float x) {
    return 0.5f * x * (1.0f + erff(x * 0.70710678118654752440f));
}

#if HAVE_ASYNC_LDS
__device__ __forceinline__ void async_cp16(const _Float16* g, _Float16* l) {
    auto gp = (__attribute__((address_space(1))) v4i_vs*)(void*)g;
    auto lp = (__attribute__((address_space(3))) v4i_vs*)(void*)l;
    __builtin_amdgcn_global_load_async_to_lds_b128(gp, lp, 0, 0);
}
template<int N>
__device__ __forceinline__ void wait_async() {
#if __has_builtin(__builtin_amdgcn_s_wait_asynccnt)
    __builtin_amdgcn_s_wait_asynccnt(N);
#else
    if constexpr (N == 0) asm volatile("s_wait_asynccnt 0x0" ::: "memory");
    else                  asm volatile("s_wait_asynccnt 0x3" ::: "memory");
#endif
}
#endif

// ---------------- utility kernels ----------------
__global__ void cvt_f32_f16(const float* __restrict__ in, _Float16* __restrict__ out, int n) {
    int i = blockIdx.x * blockDim.x + threadIdx.x;
    if (i < n) out[i] = (_Float16)in[i];
}

// weights: in f32 [K][N] -> out f16 [N][K] (transposed, so GEMM B operand is K-contiguous)
__global__ void cvt_transpose_f16(const float* __restrict__ in, _Float16* __restrict__ out,
                                  int K, int N) {
    int i = blockIdx.x * blockDim.x + threadIdx.x;
    if (i >= K * N) return;
    int k = i / N, n = i % N;
    out[(size_t)n * K + k] = (_Float16)in[i];
}

__global__ void zero_f32(float* __restrict__ p, int n) {
    int i = blockIdx.x * blockDim.x + threadIdx.x;
    if (i < n) p[i] = 0.0f;
}

// ---------------- WMMA fragment load: two 8-half runs per lane (k 8*hh.., 16+8*hh..) ----------------
// Per-lane A/B layout for V_WMMA_F32_16X16X32_F16 (ISA 7.12.2): lane = (r, hh):
// frag halves [0..7] = K[8*hh .. 8*hh+7], frag halves [8..15] = K[16+8*hh .. +7].
__device__ __forceinline__ v16h load_frag(const _Float16* rowp, int hh) {
    union { uint4 u[2]; v16h v; } tmp;
    tmp.u[0] = *reinterpret_cast<const uint4*>(rowp + 8 * hh);
    tmp.u[1] = *reinterpret_cast<const uint4*>(rowp + 16 + 8 * hh);
    return tmp.v;
}

__device__ __forceinline__ void mma_tile(const _Float16 (*As)[40], const _Float16 (*Bs)[40],
                                         v8f acc[2][2], int wm, int wn, int r, int hh) {
    v16h af[2], bf[2];
    af[0] = load_frag(As[32 * wm + r], hh);
    af[1] = load_frag(As[32 * wm + 16 + r], hh);
    bf[0] = load_frag(Bs[32 * wn + r], hh);
    bf[1] = load_frag(Bs[32 * wn + 16 + r], hh);
    #pragma unroll
    for (int mi = 0; mi < 2; ++mi)
        #pragma unroll
        for (int ni = 0; ni < 2; ++ni)
            acc[mi][ni] = __builtin_amdgcn_wmma_f32_16x16x32_f16(
                false, af[mi], false, bf[ni], (short)0, acc[mi][ni], false, false);
}

// ---------------- WMMA f16 GEMM: C = act(A @ Wt^T + bias [+ res]) ----------------
// A: MxK f16 row-major, Wt: NxK f16 row-major (pre-transposed weights), C: MxN f32.
// Block tile 128x64, 8 waves (4x2), each wave 2x2 WMMA 16x16x32 tiles.
// Double-buffered GLOBAL_LOAD_ASYNC_TO_LDS_B128 staging when available.
template<int ACT, int RES>
__global__ __launch_bounds__(256)
void gemm_wmma_f16(const _Float16* __restrict__ A, const _Float16* __restrict__ Wt,
                   const float* __restrict__ bias, const float* __restrict__ res,
                   float* __restrict__ C, int M, int N, int K) {
    __shared__ __align__(16) _Float16 As[2][128][40];  // 32 K-halves + pad
    __shared__ __align__(16) _Float16 Bs[2][64][40];
    const int t    = threadIdx.x;
    const int m0   = blockIdx.y * 128;
    const int n0   = blockIdx.x * 64;
    const int w    = t >> 5, lane = t & 31;
    const int wm   = w & 3, wn = w >> 2;
    const int r    = lane & 15, hh = lane >> 4;
    const int arow = t >> 1, acol = (t & 1) * 16;   // 128 rows x (2 x 16B)
    const int brow = t >> 2, bq   = (t & 3) * 8;    // 64 rows  x (4 x 16B)
    const _Float16* gA = &A[(size_t)(m0 + arow) * K + acol];
    const _Float16* gB = &Wt[(size_t)(n0 + brow) * K + bq];
    const int KT = K >> 5;

    v8f acc[2][2];
    v8f z = {};
    acc[0][0] = z; acc[0][1] = z; acc[1][0] = z; acc[1][1] = z;

#if HAVE_ASYNC_LDS
    // prologue: stage tile 0
    async_cp16(gA,     &As[0][arow][acol]);
    async_cp16(gA + 8, &As[0][arow][acol + 8]);
    async_cp16(gB,     &Bs[0][brow][bq]);
    for (int it = 0; it < KT; ++it) {
        const int cur = it & 1, nxt = cur ^ 1;
        if (it + 1 < KT) {
            const int kt = (it + 1) << 5;
            async_cp16(gA + kt,     &As[nxt][arow][acol]);
            async_cp16(gA + kt + 8, &As[nxt][arow][acol + 8]);
            async_cp16(gB + kt,     &Bs[nxt][brow][bq]);
            wait_async<3>();   // 3 newest in flight = next tile; current tile landed
        } else {
            wait_async<0>();
        }
        __syncthreads();
        mma_tile(As[cur], Bs[cur], acc, wm, wn, r, hh);
        __syncthreads();       // all reads of As[cur]/Bs[cur] done before it gets overwritten
    }
#else
    for (int it = 0; it < KT; ++it) {
        const int kt = it << 5;
        uint4 a0 = *reinterpret_cast<const uint4*>(gA + kt);
        uint4 a1 = *reinterpret_cast<const uint4*>(gA + kt + 8);
        uint4 b0 = *reinterpret_cast<const uint4*>(gB + kt);
        __syncthreads();
        *reinterpret_cast<uint4*>(&As[0][arow][acol])     = a0;
        *reinterpret_cast<uint4*>(&As[0][arow][acol + 8]) = a1;
        *reinterpret_cast<uint4*>(&Bs[0][brow][bq])       = b0;
        __syncthreads();
        mma_tile(As[0], Bs[0], acc, wm, wn, r, hh);
    }
#endif

    // Epilogue: C/D layout -> lane (r,hh): VGPR i holds row (i + 8*hh), col r
    #pragma unroll
    for (int mi = 0; mi < 2; ++mi) {
        #pragma unroll
        for (int ni = 0; ni < 2; ++ni) {
            int n = n0 + 32 * wn + 16 * ni + r;
            #pragma unroll
            for (int i = 0; i < 8; ++i) {
                int m = m0 + 32 * wm + 16 * mi + i + 8 * hh;
                float v = acc[mi][ni][i] + bias[n];
                if (RES) v += res[(size_t)m * N + n];
                if (ACT) v = gelu_exact(v);
                C[(size_t)m * N + n] = v;
            }
        }
    }
}

// ---------------- phase head: cos/sin(tanh(H @ W2 + b2) * pi) ----------------
__global__ __launch_bounds__(256)
void phase_head(const float* __restrict__ H, const float* __restrict__ W2,
                const float* __restrict__ b2, float* __restrict__ outc,
                float* __restrict__ outs) {
    int tok = blockIdx.x, tid = threadIdx.x;
    int p = tid & 15, g = tid >> 4;
    const float* h = H + (size_t)tok * DDIM;
    float sum = 0.f;
    for (int d = g; d < DDIM; d += 16) sum += h[d] * W2[d * 16 + p];
    __shared__ float red[256];
    red[tid] = sum;
    __syncthreads();
    for (int s = 128; s >= 16; s >>= 1) {
        if (tid < s) red[tid] += red[tid + s];
        __syncthreads();
    }
    if (tid < 16) {
        float ph = tanhf(red[tid] + b2[tid]) * PI_F;
        float sv, cv;
        __sincosf(ph, &sv, &cv);
        outc[tok * 16 + tid] = cv;
        outs[tok * 16 + tid] = sv;
    }
}

// ---------------- gate head: sigmoid(G1 @ gw2 + gb2) ----------------
__global__ __launch_bounds__(256)
void gate_head(const float* __restrict__ G1, const float* __restrict__ gw2,
               const float* __restrict__ gb2, float* __restrict__ gate) {
    int tok = blockIdx.x, tid = threadIdx.x;
    __shared__ float red[256];
    red[tid] = G1[(size_t)tok * 256 + tid] * gw2[tid];
    __syncthreads();
    for (int s = 128; s >= 1; s >>= 1) {
        if (tid < s) red[tid] += red[tid + s];
        __syncthreads();
    }
    if (tid == 0) gate[tok] = 1.f / (1.f + expf(-(red[0] + gb2[0])));
}

// ---------------- positional phasor table ----------------
__global__ void pos_phasor(const float* __restrict__ freqs, float* __restrict__ pc,
                           float* __restrict__ ps) {
    int i = blockIdx.x * blockDim.x + threadIdx.x;
    if (i >= LSEQ * 16) return;
    int l = i >> 4, p = i & 15;
    double ph = (double)l * (double)freqs[p] * 6.283185307179586476925286766559;
    pc[i] = (float)cos(ph);
    ps[i] = (float)sin(ph);
}

// ---------------- scan phase A: per-chunk complex plane sums ----------------
// grid.x = B*NCHUNK (b = x>>5, c = x&31), grid.y = 4 plane-groups of 8.
// planes 0..15 = learned set planes (kc/ks), 16..31 = positional planes (pc/ps).
__global__ __launch_bounds__(256)
void scan_chunk_sums(const float* __restrict__ V, const float* __restrict__ kc,
                     const float* __restrict__ ks, const float* __restrict__ pc,
                     const float* __restrict__ ps, float2* __restrict__ S) {
    int b = blockIdx.x >> 5, c = blockIdx.x & 31, pg = blockIdx.y;
    int tid = threadIdx.x;
    __shared__ float phc[CHUNK][8], phs[CHUNK][8];
    for (int i = tid; i < CHUNK * 8; i += 256) {
        int tok = i >> 3, pp = i & 7, gp = pg * 8 + pp;
        int l = c * CHUNK + tok, t = b * LSEQ + l;
        if (gp < 16) { phc[tok][pp] = kc[t * 16 + gp];      phs[tok][pp] = ks[t * 16 + gp]; }
        else         { phc[tok][pp] = pc[l * 16 + gp - 16]; phs[tok][pp] = ps[l * 16 + gp - 16]; }
    }
    __syncthreads();
    float sre[8][2] = {}, sim[8][2] = {};
    int d0 = tid;
    for (int tok = 0; tok < CHUNK; ++tok) {
        int t = b * LSEQ + c * CHUNK + tok;
        float v0 = V[(size_t)t * DDIM + d0];
        float v1 = V[(size_t)t * DDIM + d0 + 256];
        #pragma unroll
        for (int p = 0; p < 8; ++p) {
            float cc = phc[tok][p], ss = phs[tok][p];
            sre[p][0] += cc * v0; sim[p][0] += ss * v0;
            sre[p][1] += cc * v1; sim[p][1] += ss * v1;
        }
    }
    #pragma unroll
    for (int p = 0; p < 8; ++p) {
        size_t base = (((size_t)(b * 32 + pg * 8 + p)) * NCHUNK + c) * DDIM;
        S[base + d0]       = make_float2(sre[p][0], sim[p][0]);
        S[base + d0 + 256] = make_float2(sre[p][1], sim[p][1]);
    }
}

// ---------------- scan phase B: in-place exclusive prefix over chunks ----------------
__global__ void scan_prefix(float2* __restrict__ S) {
    int idx = blockIdx.x * blockDim.x + threadIdx.x;  // 2*32*512
    if (idx >= 2 * 32 * DDIM) return;
    int d = idx & 511, gp = (idx >> 9) & 31, b = idx >> 14;
    size_t base = ((size_t)(b * 32 + gp)) * NCHUNK * DDIM + d;
    float2 run = make_float2(0.f, 0.f);
    for (int c = 0; c < NCHUNK; ++c) {
        size_t off = base + (size_t)c * DDIM;
        float2 v = S[off];
        S[off] = run;
        run.x += v.x; run.y += v.y;
    }
}

// ---------------- scan phase C: local scan + conj-retrieve + gated blend ----------------
__global__ __launch_bounds__(256)
void scan_retrieve(const float* __restrict__ V, const float* __restrict__ kc,
                   const float* __restrict__ ks, const float* __restrict__ qc,
                   const float* __restrict__ qs, const float* __restrict__ pc,
                   const float* __restrict__ ps, const float* __restrict__ gate,
                   const float* __restrict__ set_w, const float* __restrict__ pos_w,
                   const float2* __restrict__ S, float* __restrict__ total) {
    int b = blockIdx.x >> 5, c = blockIdx.x & 31, pg = blockIdx.y;
    int tid = threadIdx.x;
    bool is_set = (pg < 2);
    __shared__ float phc[CHUNK][8], phs[CHUNK][8], rqc[CHUNK][8], rqs[CHUNK][8];
    __shared__ float gsh[CHUNK];
    for (int i = tid; i < CHUNK * 8; i += 256) {
        int tok = i >> 3, pp = i & 7, gp = pg * 8 + pp;
        int l = c * CHUNK + tok, t = b * LSEQ + l;
        if (gp < 16) {
            phc[tok][pp] = kc[t * 16 + gp]; phs[tok][pp] = ks[t * 16 + gp];
            rqc[tok][pp] = qc[t * 16 + gp]; rqs[tok][pp] = qs[t * 16 + gp];
        } else {
            float cv = pc[l * 16 + gp - 16], sv = ps[l * 16 + gp - 16];
            phc[tok][pp] = cv; phs[tok][pp] = sv;
            rqc[tok][pp] = cv; rqs[tok][pp] = sv;
        }
    }
    for (int i = tid; i < CHUNK; i += 256) gsh[i] = gate[b * LSEQ + c * CHUNK + i];
    __syncthreads();

    float e0 = expf(set_w[0]), e1 = expf(set_w[1]), e2 = expf(set_w[2]), e3 = expf(set_w[3]);
    float esum = e0 + e1 + e2 + e3;
    float wset[4] = { e0 / esum, e1 / esum, e2 / esum, e3 / esum };
    float sigpw = 1.f / (1.f + expf(-pos_w[0]));
    float wpl[8];
    #pragma unroll
    for (int p = 0; p < 8; ++p) {
        int gp = pg * 8 + p;
        wpl[p] = is_set ? wset[gp >> 2] : sigpw;
    }

    int d0 = tid;
    float mre[8][2], mim[8][2];
    #pragma unroll
    for (int p = 0; p < 8; ++p) {  // carry-in from exclusive prefix
        size_t base = (((size_t)(b * 32 + pg * 8 + p)) * NCHUNK + c) * DDIM;
        float2 s0 = S[base + d0], s1 = S[base + d0 + 256];
        mre[p][0] = s0.x; mim[p][0] = s0.y;
        mre[p][1] = s1.x; mim[p][1] = s1.y;
    }
    for (int tok = 0; tok < CHUNK; ++tok) {
        int t = b * LSEQ + c * CHUNK + tok;
        float v0 = V[(size_t)t * DDIM + d0];
        float v1 = V[(size_t)t * DDIM + d0 + 256];
        float con0 = 0.f, con1 = 0.f;
        #pragma unroll
        for (int p = 0; p < 8; ++p) {
            float cc = phc[tok][p], ss = phs[tok][p];
            mre[p][0] += cc * v0; mim[p][0] += ss * v0;   // bind (inclusive)
            mre[p][1] += cc * v1; mim[p][1] += ss * v1;
            float qcc = rqc[tok][p], qss = rqs[tok][p];
            con0 += wpl[p] * (qcc * mre[p][0] + qss * mim[p][0]);  // Re(conj(q)*mem)
            con1 += wpl[p] * (qcc * mre[p][1] + qss * mim[p][1]);
        }
        float g = gsh[tok];
        float gm = is_set ? g : (1.f - g);
        atomicAdd(&total[(size_t)t * DDIM + d0],       gm * con0);
        atomicAdd(&total[(size_t)t * DDIM + d0 + 256], gm * con1);
    }
}

// ---------------- 1/sqrt((l+1)*PLANES) scale + LayerNorm -> f16 ----------------
__global__ __launch_bounds__(256)
void norm_ln(const float* __restrict__ total, const float* __restrict__ ln_g,
             const float* __restrict__ ln_b, _Float16* __restrict__ out_h) {
    int tok = blockIdx.x, tid = threadIdx.x;
    int l = tok & (LSEQ - 1);
    float inv = rsqrtf((float)(l + 1) * 4.0f);
    float h0 = total[(size_t)tok * DDIM + tid] * inv;
    float h1 = total[(size_t)tok * DDIM + tid + 256] * inv;
    __shared__ float r1[256], r2[256];
    r1[tid] = h0 + h1;
    r2[tid] = h0 * h0 + h1 * h1;
    __syncthreads();
    for (int s = 128; s >= 1; s >>= 1) {
        if (tid < s) { r1[tid] += r1[tid + s]; r2[tid] += r2[tid + s]; }
        __syncthreads();
    }
    float mu = r1[0] * (1.f / 512.f);
    float var = r2[0] * (1.f / 512.f) - mu * mu;
    float rinv = rsqrtf(var + 1e-5f);
    out_h[(size_t)tok * DDIM + tid]       = (_Float16)((h0 - mu) * rinv * ln_g[tid] + ln_b[tid]);
    out_h[(size_t)tok * DDIM + tid + 256] = (_Float16)((h1 - mu) * rinv * ln_g[tid + 256] + ln_b[tid + 256]);
}

extern "C" void kernel_launch(void* const* d_in, const int* in_sizes, int n_in,
                              void* d_out, int out_size, void* d_ws, size_t ws_size,
                              hipStream_t stream) {
    (void)in_sizes; (void)n_in; (void)out_size; (void)ws_size;
    const float* x     = (const float*)d_in[0];
    const float* ke_w1 = (const float*)d_in[1];
    const float* ke_b1 = (const float*)d_in[2];
    const float* ke_w2 = (const float*)d_in[3];
    const float* ke_b2 = (const float*)d_in[4];
    const float* qe_w1 = (const float*)d_in[5];
    const float* qe_b1 = (const float*)d_in[6];
    const float* qe_w2 = (const float*)d_in[7];
    const float* qe_b2 = (const float*)d_in[8];
    const float* v_w   = (const float*)d_in[9];
    const float* v_b   = (const float*)d_in[10];
    const float* ln_g  = (const float*)d_in[11];
    const float* ln_b  = (const float*)d_in[12];
    const float* out_w = (const float*)d_in[13];
    const float* out_b = (const float*)d_in[14];
    const float* set_w = (const float*)d_in[15];
    const float* freqs = (const float*)d_in[16];
    const float* pos_w = (const float*)d_in[17];
    const float* gw1   = (const float*)d_in[18];
    const float* gb1   = (const float*)d_in[19];
    const float* gw2   = (const float*)d_in[20];
    const float* gb2   = (const float*)d_in[21];
    float* out = (float*)d_out;

    char* ws = (char*)d_ws;
    size_t off = 0;
    auto alloc = [&](size_t bytes) { size_t o = off; off += (bytes + 255) & ~(size_t)255; return o; };

    size_t o_xh   = alloc((size_t)TOKENS * DDIM * 2);      // x as f16; later reused for LN output f16
    size_t o_wke  = alloc(512 * 512 * 2);                  // transposed f16 weights [N][K]
    size_t o_wqe  = alloc(512 * 512 * 2);
    size_t o_wv   = alloc(512 * 512 * 2);
    size_t o_wg   = alloc(512 * 256 * 2);
    size_t o_wo   = alloc(512 * 512 * 2);
    size_t o_kq   = alloc((size_t)TOKENS * DDIM * 4 * 2);  // k_h1 + q_h1; later aliased by S buffer
    size_t o_v    = alloc((size_t)TOKENS * DDIM * 4);
    size_t o_g1   = alloc((size_t)TOKENS * 256 * 4);
    size_t o_kc   = alloc((size_t)TOKENS * 16 * 4);
    size_t o_ks   = alloc((size_t)TOKENS * 16 * 4);
    size_t o_qc   = alloc((size_t)TOKENS * 16 * 4);
    size_t o_qs   = alloc((size_t)TOKENS * 16 * 4);
    size_t o_pc   = alloc((size_t)LSEQ * 16 * 4);
    size_t o_ps   = alloc((size_t)LSEQ * 16 * 4);
    size_t o_gate = alloc((size_t)TOKENS * 4);
    size_t o_tot  = alloc((size_t)TOKENS * DDIM * 4);

    _Float16* xh   = (_Float16*)(ws + o_xh);
    _Float16* wkeh = (_Float16*)(ws + o_wke);
    _Float16* wqeh = (_Float16*)(ws + o_wqe);
    _Float16* wvh  = (_Float16*)(ws + o_wv);
    _Float16* wgh  = (_Float16*)(ws + o_wg);
    _Float16* woh  = (_Float16*)(ws + o_wo);
    float*    kh1  = (float*)(ws + o_kq);
    float*    qh1  = (float*)(ws + o_kq + (size_t)TOKENS * DDIM * 4);
    float2*   Sbuf = (float2*)(ws + o_kq);                 // alias: k_h1/q_h1 dead by then
    float*    Vb   = (float*)(ws + o_v);
    float*    g1   = (float*)(ws + o_g1);
    float*    kcb  = (float*)(ws + o_kc);
    float*    ksb  = (float*)(ws + o_ks);
    float*    qcb  = (float*)(ws + o_qc);
    float*    qsb  = (float*)(ws + o_qs);
    float*    pcb  = (float*)(ws + o_pc);
    float*    psb  = (float*)(ws + o_ps);
    float*    gtb  = (float*)(ws + o_gate);
    float*    totb = (float*)(ws + o_tot);
    _Float16* hln  = (_Float16*)(ws + o_xh);               // alias: x_f16 dead by then

    // 1) converts (+ weight transpose so B operand is K-contiguous)
    cvt_f32_f16<<<(TOKENS * DDIM + 255) / 256, 256, 0, stream>>>(x, xh, TOKENS * DDIM);
    cvt_transpose_f16<<<(512 * 512 + 255) / 256, 256, 0, stream>>>(ke_w1, wkeh, 512, 512);
    cvt_transpose_f16<<<(512 * 512 + 255) / 256, 256, 0, stream>>>(qe_w1, wqeh, 512, 512);
    cvt_transpose_f16<<<(512 * 512 + 255) / 256, 256, 0, stream>>>(v_w,   wvh,  512, 512);
    cvt_transpose_f16<<<(512 * 256 + 255) / 256, 256, 0, stream>>>(gw1,   wgh,  512, 256);
    cvt_transpose_f16<<<(512 * 512 + 255) / 256, 256, 0, stream>>>(out_w, woh,  512, 512);

    // 2) big WMMA GEMMs
    dim3 g512(512 / 64, TOKENS / 128), g256(256 / 64, TOKENS / 128);
    gemm_wmma_f16<1, 0><<<g512, 256, 0, stream>>>(xh, wkeh, ke_b1, nullptr, kh1, TOKENS, 512, 512);
    gemm_wmma_f16<1, 0><<<g512, 256, 0, stream>>>(xh, wqeh, qe_b1, nullptr, qh1, TOKENS, 512, 512);
    gemm_wmma_f16<0, 0><<<g512, 256, 0, stream>>>(xh, wvh,  v_b,   nullptr, Vb,  TOKENS, 512, 512);
    gemm_wmma_f16<1, 0><<<g256, 256, 0, stream>>>(xh, wgh,  gb1,   nullptr, g1,  TOKENS, 256, 512);

    // 3) heads + positional phasors
    phase_head<<<TOKENS, 256, 0, stream>>>(kh1, ke_w2, ke_b2, kcb, ksb);
    phase_head<<<TOKENS, 256, 0, stream>>>(qh1, qe_w2, qe_b2, qcb, qsb);
    gate_head<<<TOKENS, 256, 0, stream>>>(g1, gw2, gb2, gtb);
    pos_phasor<<<(LSEQ * 16 + 255) / 256, 256, 0, stream>>>(freqs, pcb, psb);

    // 4) chunked complex scan (k_h1/q_h1 region is now dead -> S buffer)
    dim3 gscan(2 * NCHUNK, 4);
    scan_chunk_sums<<<gscan, 256, 0, stream>>>(Vb, kcb, ksb, pcb, psb, Sbuf);
    scan_prefix<<<(2 * 32 * DDIM + 255) / 256, 256, 0, stream>>>(Sbuf);
    zero_f32<<<(TOKENS * DDIM + 255) / 256, 256, 0, stream>>>(totb, TOKENS * DDIM);
    scan_retrieve<<<gscan, 256, 0, stream>>>(Vb, kcb, ksb, qcb, qsb, pcb, psb,
                                             gtb, set_w, pos_w, Sbuf, totb);

    // 5) scale + LayerNorm -> f16 (x_f16 region is now dead -> hln)
    norm_ln<<<TOKENS, 256, 0, stream>>>(totb, ln_g, ln_b, hln);

    // 6) output projection with residual, straight into d_out
    gemm_wmma_f16<0, 1><<<g512, 256, 0, stream>>>(hln, woh, out_b, x, out, TOKENS, 512, 512);
}
